// RetinaFace_Biubug6_12051678233266
// MI455X (gfx1250) — matrix operand order repaired and checked by reading, not compiled, not run on hardware
//
#include <hip/hip_runtime.h>

// ---------------------------------------------------------------------------
// RetinaFace head on gfx1250 (CDNA5, wave32, WMMA).
// 3x3 convs = implicit GEMM on v_wmma_f32_16x16x32_f16 (f16 in, f32 accum).
// K-dimension is TAP-MAJOR (k = tap*Cin + ci) so every 32-wide K-step lives
// inside one tap: no integer div/rem in the hot loop, scalar boundary tests.
// Each wave holds 4 accumulators (16 cout x 64 pixels) to amortize A loads.
// ---------------------------------------------------------------------------

typedef _Float16 v8h  __attribute__((ext_vector_type(8)));
typedef _Float16 v16h __attribute__((ext_vector_type(16)));
typedef float    v8f  __attribute__((ext_vector_type(8)));

#define NCAND    16800
#define KTOP     512
#define SCORETHR 0.02f
#define IOUTHR   0.4f
#define BS_STRIDE 40   // LDS row stride in halfs: 80B (16B aligned, bank-spread)

// ---------------- weight repack: OIHW f32 -> [Cout][tap*Cin+ci] f16 ----------
__global__ void repack_w(const float* __restrict__ in, _Float16* __restrict__ out,
                         int Cout, int Cin) {
  int i = blockIdx.x * blockDim.x + threadIdx.x;
  int total = Cout * Cin * 9;
  if (i >= total) return;
  int tap = i % 9;
  int t   = i / 9;
  int ci  = t % Cin;
  int co  = t / Cin;
  out[(size_t)co * Cin * 9 + tap * Cin + ci] = (_Float16)in[i];
}

// ---------------- 3x3 conv, pad=1, implicit GEMM via WMMA ----------------
// X  : [B, Cin, H, W]  (TIn = float for net inputs, _Float16 intermediates)
// Wk : [Cout, 9*Cin] f16, tap-major K
// Y  : [B, CoutTot, H, W] f16 at channel offset coOff, fused bias+ReLU.
// Block = (Cout/16) waves, 64-pixel tile; wave w owns cout rows [16w,16w+16)
// and all 4 pixel subtiles (4 wmma accumulators).
template <typename TIn>
__global__ void conv3x3_wmma(const TIn* __restrict__ X,
                             const _Float16* __restrict__ Wk,
                             const float* __restrict__ bias,
                             _Float16* __restrict__ Y,
                             int Cin, int H, int W,
                             int coOff, int CoutTot)
{
  const int HW    = H * W;
  const int K     = Cin * 9;
  const int pbase = blockIdx.x * 64;
  const int b     = blockIdx.y;
  const int tid   = threadIdx.x;
  const int wave  = tid >> 5;
  const int lane  = tid & 31;
  const int half  = lane >> 4;          // lane group 0-15 vs 16-31
  const int ln16  = lane & 15;          // A row m / B,D column n
  const int m0    = wave * 16;

  const TIn* Xb = X + (size_t)b * Cin * HW;

  __shared__ alignas(16) _Float16 Bs[64 * BS_STRIDE];   // [pixel][k_local]

  v8f acc[4];
  #pragma unroll
  for (int t = 0; t < 4; ++t) acc[t] = (v8f){0.f,0.f,0.f,0.f,0.f,0.f,0.f,0.f};

  // ---- fill-thread fixed assignment (pixel fixed per thread) ----
  const int  nn    = tid & 63;               // pixel within tile
  const int  klq   = tid >> 6;               // first k_local
  const int  kstep = blockDim.x >> 6;        // 4 (256 thr) or 2 (128 thr)
  const int  p     = pbase + nn;
  const int  py    = p / W, px = p - py * W; // hoisted: once per kernel
  const bool pin   = (p < HW);

  // A fragment: elements 0..7 at k0+8*half, 8..15 at k0+8*half+16
  const _Float16* wrow0 = Wk + (size_t)(m0 + ln16) * K + half * 8;

  for (int tap = 0; tap < 9; ++tap) {
    const int  ky = tap / 3, kx = tap - ky * 3;     // tiny, outside hot loop
    const int  iy = py + ky - 1, ix = px + kx - 1;
    const bool valid = pin && (iy >= 0) && (iy < H) && (ix >= 0) && (ix < W);
    const TIn* grow  = Xb + (size_t)iy * W + ix;    // + ci*HW per element

    for (int cb = 0; cb < Cin; cb += 32) {
      // ---- cooperative fill: Bs[nn][kl] = X[(cb+kl)*HW + iy*W + ix] ----
      {
        const TIn* g2 = grow + (size_t)(cb + klq) * HW;
        #pragma unroll 2
        for (int kl = klq; kl < 32; kl += kstep) {
          float v = valid ? (float)(*g2) : 0.0f;
          Bs[nn * BS_STRIDE + kl] = (_Float16)v;
          g2 += (size_t)kstep * HW;
        }
      }
      __syncthreads();

      // ---- one A fragment, four WMMAs ----
      const _Float16* wrow = wrow0 + tap * Cin + cb;
      __builtin_prefetch(wrow + 32, 0, 0);          // global_prefetch_b8
      v8h alo = *(const v8h*)(wrow);
      v8h ahi = *(const v8h*)(wrow + 16);
      v16h amat = __builtin_shufflevector(alo, ahi,
                      0,1,2,3,4,5,6,7,8,9,10,11,12,13,14,15);

      #pragma unroll
      for (int t = 0; t < 4; ++t) {
        // B fragment: elements 0..7 at K = 8*half, 8..15 at K = 8*half+16 (16B aligned)
        const v8h* bp = (const v8h*)&Bs[(t * 16 + ln16) * BS_STRIDE + half * 8];
        v16h bmat = __builtin_shufflevector(bp[0], bp[2],
                        0,1,2,3,4,5,6,7,8,9,10,11,12,13,14,15);
        acc[t] = __builtin_amdgcn_wmma_f32_16x16x32_f16(false, amat, false, bmat,
                                                        (short)0, acc[t],
                                                        false, false);
      }
      __syncthreads();
    }
  }

  // ---- epilogue: D element r -> m = m0 + r + 8*half, n = lane&15 ----
  float bs[8];
  #pragma unroll
  for (int r = 0; r < 8; ++r) bs[r] = bias[m0 + r + half * 8];

  #pragma unroll
  for (int t = 0; t < 4; ++t) {
    int pp = pbase + t * 16 + ln16;
    if (pp < HW) {
      #pragma unroll
      for (int r = 0; r < 8; ++r) {
        int co = m0 + r + half * 8;
        float v = acc[t][r] + bs[r];
        v = fmaxf(v, 0.0f);   // SSH trailing relu(concat) fused per branch
        Y[((size_t)b * CoutTot + coOff + co) * HW + pp] = (_Float16)v;
      }
    }
  }
}

// ---------------- fused 1x1 heads + softmax + prior decode ----------------
__global__ void head_decode(const _Float16* __restrict__ F,
                            const float* __restrict__ clsW, const float* __restrict__ clsB,
                            const float* __restrict__ boxW, const float* __restrict__ boxB,
                            float* __restrict__ boxesOut, float* __restrict__ scrOut,
                            int H, int W, int levelOff, float strideF,
                            float ms0, float ms1)
{
  const int HW  = H * W;
  const int b   = blockIdx.y;
  const int tid = threadIdx.x;

  __shared__ float ws[12 * 256];            // cls (4x256) then box (8x256)
  for (int idx = tid; idx < 4 * 256; idx += blockDim.x) ws[idx]        = clsW[idx];
  for (int idx = tid; idx < 8 * 256; idx += blockDim.x) ws[1024 + idx] = boxW[idx];
  __syncthreads();

  const int p = blockIdx.x * blockDim.x + tid;
  if (p >= HW) return;

  const _Float16* Fb = F + (size_t)b * 256 * HW + p;
  float acc[12];
  #pragma unroll
  for (int c = 0; c < 12; ++c) acc[c] = 0.f;
  for (int ci = 0; ci < 256; ++ci) {
    float x = (float)Fb[(size_t)ci * HW];
    #pragma unroll
    for (int c = 0; c < 4; ++c) acc[c]     += ws[c * 256 + ci] * x;
    #pragma unroll
    for (int c = 0; c < 8; ++c) acc[4 + c] += ws[1024 + c * 256 + ci] * x;
  }

  const int   py  = p / W, px = p - py * W;
  const float pcx = (px + 0.5f) * strideF * (1.0f / 640.0f);
  const float pcy = (py + 0.5f) * strideF * (1.0f / 640.0f);
  const float msz[2] = {ms0, ms1};

  #pragma unroll
  for (int a = 0; a < 2; ++a) {
    float c0 = acc[a * 2 + 0] + clsB[a * 2 + 0];
    float c1 = acc[a * 2 + 1] + clsB[a * 2 + 1];
    float score = 1.0f / (1.0f + __expf(c0 - c1));   // softmax[...,1]

    float r0 = acc[4 + a * 4 + 0] + boxB[a * 4 + 0];
    float r1 = acc[4 + a * 4 + 1] + boxB[a * 4 + 1];
    float r2 = acc[4 + a * 4 + 2] + boxB[a * 4 + 2];
    float r3 = acc[4 + a * 4 + 3] + boxB[a * 4 + 3];

    float pw = msz[a], ph = msz[a];
    float cx = pcx + r0 * 0.1f * pw;
    float cy = pcy + r1 * 0.1f * ph;
    float w  = pw * __expf(r2 * 0.2f);
    float h  = ph * __expf(r3 * 0.2f);
    float tlx = cx - 0.5f * w, tly = cy - 0.5f * h;

    size_t ni = (size_t)b * NCAND + levelOff + p * 2 + a;
    boxesOut[ni * 4 + 0] = tlx;
    boxesOut[ni * 4 + 1] = tly;
    boxesOut[ni * 4 + 2] = tlx + w;
    boxesOut[ni * 4 + 3] = tly + h;
    scrOut[ni] = score;
  }
}

// ---------------- exact top-K via rank counting ----------------
// Reproduces lax.top_k (descending, ties by lower index): ranks permute
// 0..N-1, so output slots 0..511 are each written exactly once.
__global__ void topk_rank(const float* __restrict__ scr, const float* __restrict__ boxes,
                          float* __restrict__ outB, float* __restrict__ outS)
{
  const int b   = blockIdx.y;
  const int tid = threadIdx.x;
  const int i   = blockIdx.x * blockDim.x + tid;
  const float* srow = scr + (size_t)b * NCAND;

  float si = -2.0f;
  if (i < NCAND) {
    float s = srow[i];
    si = (s >= SCORETHR) ? s : -1.0f;
  }

  int rank = 0;
  __shared__ float sc[256];
  for (int j0 = 0; j0 < NCAND; j0 += 256) {
    __syncthreads();
    if (j0 + tid < NCAND) sc[tid] = srow[j0 + tid];
    __syncthreads();
    int lim = NCAND - j0; if (lim > 256) lim = 256;
    for (int jj = 0; jj < lim; ++jj) {
      float s  = sc[jj];
      float sj = (s >= SCORETHR) ? s : -1.0f;
      int   j  = j0 + jj;
      rank += ((sj > si) || (sj == si && j < i)) ? 1 : 0;
    }
  }

  if (i < NCAND && rank < KTOP) {
    size_t o   = (size_t)b * KTOP + rank;
    size_t src = ((size_t)b * NCAND + i) * 4;
    outS[o]         = si;
    outB[o * 4 + 0] = boxes[src + 0];
    outB[o * 4 + 1] = boxes[src + 1];
    outB[o * 4 + 2] = boxes[src + 2];
    outB[o * 4 + 3] = boxes[src + 3];
  }
}

// ---------------- greedy NMS, one 512-thread block per image ----------------
__global__ void nms_kernel(const float* __restrict__ topB, const float* __restrict__ topS,
                           float* __restrict__ keepOut)
{
  const int b = blockIdx.x;
  const int j = threadIdx.x;
  __shared__ float x0s[KTOP], y0s[KTOP], x1s[KTOP], y1s[KTOP], ars[KTOP];
  __shared__ int   keep[KTOP];

  const float* bp = topB + ((size_t)b * KTOP + j) * 4;
  float x0 = bp[0], y0 = bp[1], x1 = bp[2], y1 = bp[3];
  x0s[j] = x0; y0s[j] = y0; x1s[j] = x1; y1s[j] = y1;
  float aj = fmaxf(x1 - x0, 0.f) * fmaxf(y1 - y0, 0.f);
  ars[j] = aj;
  keep[j] = (topS[(size_t)b * KTOP + j] > 0.0f) ? 1 : 0;
  __syncthreads();

  for (int i = 0; i < KTOP; ++i) {
    if (keep[i] && j > i) {
      float ix0 = fmaxf(x0s[i], x0);
      float iy0 = fmaxf(y0s[i], y0);
      float ix1 = fminf(x1s[i], x1);
      float iy1 = fminf(y1s[i], y1);
      float iw = fmaxf(ix1 - ix0, 0.f), ih = fmaxf(iy1 - iy0, 0.f);
      float inter = iw * ih;
      float iou = inter / (ars[i] + aj - inter + 1e-9f);
      if (iou > IOUTHR) keep[j] = 0;
    }
    __syncthreads();
  }
  keepOut[(size_t)b * KTOP + j] = keep[j] ? 1.0f : 0.0f;
}

// ---------------------------------------------------------------------------
extern "C" void kernel_launch(void* const* d_in, const int* in_sizes, int n_in,
                              void* d_out, int out_size, void* d_ws, size_t ws_size,
                              hipStream_t stream)
{
  (void)in_sizes; (void)n_in; (void)out_size; (void)ws_size;
  const int   B = 8;
  const int   Hs[3]      = {80, 40, 20};
  const float strides[3] = {8.f, 16.f, 32.f};
  const float ms[3][2]   = {{16.f/640.f, 32.f/640.f},
                            {64.f/640.f, 128.f/640.f},
                            {256.f/640.f, 512.f/640.f}};
  const int   levOff[3]  = {0, 12800, 16000};

  const float* x[3] = {(const float*)d_in[0], (const float*)d_in[1], (const float*)d_in[2]};
  const float *wsrc[3][5], *bsrc[3][5], *clsW[3], *clsB[3], *boxW[3], *boxB[3];
  for (int l = 0; l < 3; ++l) {
    int base = 3 + l * 14;   // ssh{l}: w1,b1,...,w5,b5 then cls_w,cls_b,box_w,box_b
    for (int k = 0; k < 5; ++k) {
      wsrc[l][k] = (const float*)d_in[base + 2 * k + 0];
      bsrc[l][k] = (const float*)d_in[base + 2 * k + 1];
    }
    clsW[l] = (const float*)d_in[base + 10];
    clsB[l] = (const float*)d_in[base + 11];
    boxW[l] = (const float*)d_in[base + 12];
    boxB[l] = (const float*)d_in[base + 13];
  }

  // ---- workspace carve-up ----
  char*  ws  = (char*)d_ws;
  size_t cur = 0;
  auto carve = [&](size_t bytes) -> void* {
    void* p = ws + cur;
    cur = (cur + bytes + 255) & ~(size_t)255;
    return p;
  };

  const int wCout[5] = {128, 64, 64, 64, 64};
  const int wCin[5]  = {256, 256, 64, 64, 64};
  _Float16* w16[3][5];
  for (int l = 0; l < 3; ++l)
    for (int k = 0; k < 5; ++k)
      w16[l][k] = (_Float16*)carve((size_t)wCout[k] * wCin[k] * 9 * sizeof(_Float16));

  _Float16 *tBuf[3], *uBuf[3], *Fbuf[3];
  for (int l = 0; l < 3; ++l) {
    size_t HW = (size_t)Hs[l] * Hs[l];
    tBuf[l] = (_Float16*)carve((size_t)B * 64  * HW * sizeof(_Float16));
    uBuf[l] = (_Float16*)carve((size_t)B * 64  * HW * sizeof(_Float16));
    Fbuf[l] = (_Float16*)carve((size_t)B * 256 * HW * sizeof(_Float16));
  }
  float* boxes = (float*)carve((size_t)B * NCAND * 4 * sizeof(float));
  float* scr   = (float*)carve((size_t)B * NCAND * sizeof(float));

  // ---- repack conv weights to f16, tap-major K ----
  for (int l = 0; l < 3; ++l)
    for (int k = 0; k < 5; ++k) {
      int n = wCout[k] * wCin[k] * 9;
      repack_w<<<dim3((n + 255) / 256), dim3(256), 0, stream>>>(
          wsrc[l][k], w16[l][k], wCout[k], wCin[k]);
    }

  // ---- SSH per level: 5 WMMA convs (final relu fused into each branch) ----
  for (int l = 0; l < 3; ++l) {
    int H = Hs[l], W = Hs[l], HW = H * W;
    dim3 g((HW + 63) / 64, B);
    // y1 = relu(conv(x, w1))  -> F[:, 0:128]
    conv3x3_wmma<float><<<g, 256, 0, stream>>>(x[l], w16[l][0], bsrc[l][0], Fbuf[l], 256, H, W, 0,   256);
    // t  = relu(conv(x, w2))  -> 64ch
    conv3x3_wmma<float><<<g, 128, 0, stream>>>(x[l], w16[l][1], bsrc[l][1], tBuf[l], 256, H, W, 0,   64);
    // y2 = relu(conv(t, w3))  -> F[:, 128:192]
    conv3x3_wmma<_Float16><<<g, 128, 0, stream>>>(tBuf[l], w16[l][2], bsrc[l][2], Fbuf[l], 64, H, W, 128, 256);
    // u  = relu(conv(t, w4))  -> 64ch
    conv3x3_wmma<_Float16><<<g, 128, 0, stream>>>(tBuf[l], w16[l][3], bsrc[l][3], uBuf[l], 64, H, W, 0,   64);
    // y3 = relu(conv(u, w5))  -> F[:, 192:256]
    conv3x3_wmma<_Float16><<<g, 128, 0, stream>>>(uBuf[l], w16[l][4], bsrc[l][4], Fbuf[l], 64, H, W, 192, 256);

    dim3 gh((HW + 255) / 256, B);
    head_decode<<<gh, 256, 0, stream>>>(Fbuf[l], clsW[l], clsB[l], boxW[l], boxB[l],
                                        boxes, scr, H, W, levOff[l], strides[l],
                                        ms[l][0], ms[l][1]);
  }

  // ---- outputs: top_b [8,512,4] | top_s [8,512] | keep [8,512] ----
  float* outB = (float*)d_out;
  float* outS = outB + (size_t)B * KTOP * 4;
  float* outK = outS + (size_t)B * KTOP;

  dim3 gt((NCAND + 255) / 256, B);
  topk_rank<<<gt, 256, 0, stream>>>(scr, boxes, outB, outS);
  nms_kernel<<<dim3(B), 512, 0, stream>>>(outB, outS, outK);
}
